// Block_75033078661864
// MI455X (gfx1250) — matrix operand order repaired and checked by reading, not compiled, not run on hardware
//
#include <hip/hip_runtime.h>
#include <hip/hip_bf16.h>
#include <math.h>

// Problem constants (ViT-L/14-style encoder block)
#define BB    32
#define SS    577
#define DD    768
#define HH    12
#define DHH   64
#define MLPD  3072
#define SPAD  608      // 19 * 32, padded sequence for WMMA tiling
#define STILES 37      // ceil(577/16)
#define MROWS (BB * SS)   // 18464 = 16 * 1154 flattened MLP rows
#define MT64  289         // ceil(18464 / 64) M-blocks for the MLP
#define LNEPS 1e-5f

typedef __attribute__((ext_vector_type(8)))  __bf16 bf16x8;
typedef __attribute__((ext_vector_type(16))) __bf16 bf16x16;
typedef __attribute__((ext_vector_type(8)))  float  f32x8;

// ---------------------------------------------------------------------------
// WMMA wrapper: D = A(16x32 bf16) * B(32x16 bf16) + C(16x16 f32)
// ---------------------------------------------------------------------------
__device__ __forceinline__ f32x8 wmma_bf16(bf16x16 a, bf16x16 b, f32x8 c) {
  return __builtin_amdgcn_wmma_f32_16x16x32_bf16(
      /*neg_a=*/false, a, /*neg_b=*/false, b,
      /*c_mod=*/(short)0, c, /*reuse_a=*/false, /*reuse_b=*/false);
}

// A-operand loader from a row-major bf16 matrix (row stride = ld elements).
// ISA 16-bit A 16x32 layout: lanes 0-15 -> row=lane, K={0..7,16..23};
// lanes 16-31 -> row=lane-16, K={8..15,24..31}. Two contiguous 16B loads.
__device__ __forceinline__ bf16x16 load_tile_a(const __bf16* p, int ld) {
  const int lane = threadIdx.x & 31;
  const int row  = lane & 15;
  const int ko   = (lane & 16) >> 1;        // 0 or 8
  const __bf16* q = p + (size_t)row * ld + ko;
  union { bf16x16 v; bf16x8 h[2]; } u;
  u.h[0] = *(const bf16x8*)(q);
  u.h[1] = *(const bf16x8*)(q + 16);
  return u.v;
}

// B-operand loader. Source must be "N-major": row n holds K contiguously
// (i.e. the transpose of the mathematical B). Same K order as A: lanes
// 0-15 -> col=lane, K={0..7,16..23}; lanes 16-31 -> col=lane-16, K={8..15,24..31}.
// Two contiguous 16B loads per lane.
__device__ __forceinline__ bf16x16 load_tile_b(const __bf16* p, int ld) {
  const int lane = threadIdx.x & 31;
  const int n    = lane & 15;
  const int ko   = (lane & 16) >> 1;        // 0 or 8
  const __bf16* q = p + (size_t)n * ld + ko;
  union { bf16x16 v; bf16x8 h[2]; } u;
  u.h[0] = *(const bf16x8*)(q);
  u.h[1] = *(const bf16x8*)(q + 16);
  return u.v;
}

// ---------------------------------------------------------------------------
// Generic batched transpose + f32 -> bf16 convert: out[b][c][r] = in[b][r][c]
// ---------------------------------------------------------------------------
__global__ void transpose_bf16_kernel(const float* __restrict__ in,
                                      __bf16* __restrict__ out, int R, int C) {
  const size_t base = (size_t)blockIdx.y * R * C;
  const int idx = blockIdx.x * 256 + threadIdx.x;
  if (idx < R * C) {
    const int r = idx / C, c = idx % C;
    out[base + (size_t)c * R + r] = (__bf16)in[base + idx];
  }
}

// ---------------------------------------------------------------------------
// LayerNorm1: x (f32, [B,S,D]) -> xn (bf16, padded [B,SPAD,D])
// One 256-thread block per row; 768 = 3 * 256.
// ---------------------------------------------------------------------------
__global__ void ln1_kernel(const float* __restrict__ x,
                           const float* __restrict__ g,
                           const float* __restrict__ be,
                           __bf16* __restrict__ xn) {
  const int row = blockIdx.x;               // 0 .. B*S-1
  const int b = row / SS, s = row % SS;
  const float* xr = x + (size_t)row * DD;
  __shared__ float red[256];
  const int tid = threadIdx.x;
  float v[3];
  float sum = 0.f;
#pragma unroll
  for (int j = 0; j < 3; ++j) { v[j] = xr[tid + j * 256]; sum += v[j]; }
  red[tid] = sum; __syncthreads();
  for (int st = 128; st > 0; st >>= 1) {
    if (tid < st) red[tid] += red[tid + st];
    __syncthreads();
  }
  const float mean = red[0] * (1.f / DD);
  __syncthreads();
  float vs = 0.f;
#pragma unroll
  for (int j = 0; j < 3; ++j) { const float d = v[j] - mean; vs += d * d; }
  red[tid] = vs; __syncthreads();
  for (int st = 128; st > 0; st >>= 1) {
    if (tid < st) red[tid] += red[tid + st];
    __syncthreads();
  }
  const float rstd = rsqrtf(red[0] * (1.f / DD) + LNEPS);
  __bf16* orow = xn + ((size_t)b * SPAD + s) * DD;
#pragma unroll
  for (int j = 0; j < 3; ++j) {
    const int i = tid + j * 256;
    orow[i] = (__bf16)((v[j] - mean) * rstd * g[i] + be[i]);
  }
}

// ---------------------------------------------------------------------------
// Residual + LayerNorm2: out = x + o (f32), yn = LN(out) (bf16, flat [B*S,D])
// ---------------------------------------------------------------------------
__global__ void resid_ln2_kernel(const float* __restrict__ x,
                                 const float* __restrict__ ob,
                                 const float* __restrict__ g,
                                 const float* __restrict__ be,
                                 float* __restrict__ outb,
                                 __bf16* __restrict__ yn) {
  const int row = blockIdx.x;
  const float* xr  = x  + (size_t)row * DD;
  const float* orr = ob + (size_t)row * DD;
  float*  outr = outb + (size_t)row * DD;
  __bf16* ynr  = yn   + (size_t)row * DD;
  __shared__ float red[256];
  const int tid = threadIdx.x;
  float v[3];
  float sum = 0.f;
#pragma unroll
  for (int j = 0; j < 3; ++j) {
    const int i = tid + j * 256;
    v[j] = xr[i] + orr[i];
    outr[i] = v[j];
    sum += v[j];
  }
  red[tid] = sum; __syncthreads();
  for (int st = 128; st > 0; st >>= 1) {
    if (tid < st) red[tid] += red[tid + st];
    __syncthreads();
  }
  const float mean = red[0] * (1.f / DD);
  __syncthreads();
  float vs = 0.f;
#pragma unroll
  for (int j = 0; j < 3; ++j) { const float d = v[j] - mean; vs += d * d; }
  red[tid] = vs; __syncthreads();
  for (int st = 128; st > 0; st >>= 1) {
    if (tid < st) red[tid] += red[tid + st];
    __syncthreads();
  }
  const float rstd = rsqrtf(red[0] * (1.f / DD) + LNEPS);
#pragma unroll
  for (int j = 0; j < 3; ++j) {
    const int i = tid + j * 256;
    ynr[i] = (__bf16)((v[j] - mean) * rstd * g[i] + be[i]);
  }
}

// ---------------------------------------------------------------------------
// QKV: per (b,h,stile) wave computes a 16x64 slab of Q, K and V.
// Q,K stored row-major [B,H,SPAD,DH]; V stored transposed Vt [B,H,DH,SPAD]
// (so the attention P*V B-operand becomes two contiguous 16B loads).
// ---------------------------------------------------------------------------
__global__ __launch_bounds__(32) void qkv_kernel(
    const __bf16* __restrict__ xn,
    const __bf16* __restrict__ wqT, const __bf16* __restrict__ wkT,
    const __bf16* __restrict__ wvT,
    const float* __restrict__ bq, const float* __restrict__ bk,
    const float* __restrict__ bv,
    __bf16* __restrict__ Q, __bf16* __restrict__ K, __bf16* __restrict__ Vt) {
  const int st = blockIdx.x, h = blockIdx.y, b = blockIdx.z;
  const int lane = threadIdx.x & 31;
  const int n  = lane & 15;
  const int g8 = (lane & 16) >> 1;          // C-tile row base: 0 or 8

  const __bf16* xrow = xn + ((size_t)b * SPAD + (size_t)st * 16) * DD + h * DHH;
  const bf16x16 a0 = load_tile_a(xrow, DD);
  const bf16x16 a1 = load_tile_a(xrow + 32, DD);
  const size_t bh = (size_t)(b * HH + h);

  const __bf16* Ws[3] = { wqT + (size_t)h * DHH * DHH,
                          wkT + (size_t)h * DHH * DHH,
                          wvT + (size_t)h * DHH * DHH };
  const float* Bs[3] = { bq + h * DHH, bk + h * DHH, bv + h * DHH };

#pragma unroll
  for (int w = 0; w < 3; ++w) {
#pragma unroll
    for (int t = 0; t < 4; ++t) {
      const bf16x16 bt0 = load_tile_b(Ws[w] + (size_t)(t * 16) * DHH, DHH);
      const bf16x16 bt1 = load_tile_b(Ws[w] + (size_t)(t * 16) * DHH + 32, DHH);
      f32x8 acc = {};
      acc = wmma_bf16(a0, bt0, acc);
      acc = wmma_bf16(a1, bt1, acc);
      const float bias = Bs[w][t * 16 + n];
      if (w == 2) {
        // Transposed V store: 8 contiguous bf16 per lane.
        bf16x8 pk;
#pragma unroll
        for (int i = 0; i < 8; ++i) pk[i] = (__bf16)(acc[i] + bias);
        *(bf16x8*)(Vt + (bh * DHH + t * 16 + n) * SPAD + (size_t)st * 16 + g8) = pk;
      } else {
        __bf16* dst = (w == 0 ? Q : K) +
                      (bh * SPAD + (size_t)st * 16) * DHH + t * 16 + n;
#pragma unroll
        for (int i = 0; i < 8; ++i)
          dst[(size_t)(g8 + i) * DHH] = (__bf16)(acc[i] + bias);
      }
    }
  }
}

// ---------------------------------------------------------------------------
// Flash attention: one wave per (b,h,16-query tile), 32 keys per iteration.
// Online softmax; P goes through a 1KB LDS tile to re-shape C-layout -> A-layout.
// ---------------------------------------------------------------------------
__global__ __launch_bounds__(32) void attn_kernel(
    const __bf16* __restrict__ Q, const __bf16* __restrict__ K,
    const __bf16* __restrict__ Vt, float* __restrict__ obuf) {
  const int qt = blockIdx.x, h = blockIdx.y, b = blockIdx.z;
  const int lane = threadIdx.x & 31;
  const int n  = lane & 15;
  const int g8 = (lane & 16) >> 1;

  const size_t bh = (size_t)(b * HH + h);
  const __bf16* Qbh = Q  + bh * SPAD * DHH;
  const __bf16* Kbh = K  + bh * SPAD * DHH;
  const __bf16* Vbh = Vt + bh * DHH * SPAD;

  const bf16x16 aq0 = load_tile_a(Qbh + (size_t)qt * 16 * DHH, DHH);
  const bf16x16 aq1 = load_tile_a(Qbh + (size_t)qt * 16 * DHH + 32, DHH);

  f32x8 o[4] = {};
  float mrow[8], lrow[8];
#pragma unroll
  for (int i = 0; i < 8; ++i) { mrow[i] = -3.0e38f; lrow[i] = 0.f; }

  __shared__ __align__(32) __bf16 Pt[16 * 32];

  for (int jp = 0; jp < SPAD / 32; ++jp) {   // 19 iterations * 32 keys
    const int kb0 = jp * 32, kb1 = jp * 32 + 16;
    f32x8 s0 = {}, s1 = {};
    {
      const bf16x16 b00 = load_tile_b(Kbh + (size_t)kb0 * DHH, DHH);
      const bf16x16 b01 = load_tile_b(Kbh + (size_t)kb0 * DHH + 32, DHH);
      s0 = wmma_bf16(aq0, b00, s0);
      s0 = wmma_bf16(aq1, b01, s0);
      const bf16x16 b10 = load_tile_b(Kbh + (size_t)kb1 * DHH, DHH);
      const bf16x16 b11 = load_tile_b(Kbh + (size_t)kb1 * DHH + 32, DHH);
      s1 = wmma_bf16(aq0, b10, s1);
      s1 = wmma_bf16(aq1, b11, s1);
    }
    const bool msk0 = (kb0 + n) >= SS;
    const bool msk1 = (kb1 + n) >= SS;
    float alpha[8];
#pragma unroll
    for (int i = 0; i < 8; ++i) {
      const float v0 = msk0 ? -3.0e38f : s0[i] * 0.125f;   // 1/sqrt(64)
      const float v1 = msk1 ? -3.0e38f : s1[i] * 0.125f;
      // row max across the 16-lane half (wave32: xor masks 1,2,4,8)
      float r = fmaxf(v0, v1);
      r = fmaxf(r, __shfl_xor(r, 1, 32));
      r = fmaxf(r, __shfl_xor(r, 2, 32));
      r = fmaxf(r, __shfl_xor(r, 4, 32));
      r = fmaxf(r, __shfl_xor(r, 8, 32));
      const float mn = fmaxf(mrow[i], r);
      alpha[i] = expf(mrow[i] - mn);
      mrow[i] = mn;
      const float p0 = expf(v0 - mn);
      const float p1 = expf(v1 - mn);
      float rs = p0 + p1;
      rs += __shfl_xor(rs, 1, 32);
      rs += __shfl_xor(rs, 2, 32);
      rs += __shfl_xor(rs, 4, 32);
      rs += __shfl_xor(rs, 8, 32);
      lrow[i] = lrow[i] * alpha[i] + rs;
      const int m = g8 + i;
      Pt[m * 32 + n]      = (__bf16)p0;      // ds_store
      Pt[m * 32 + 16 + n] = (__bf16)p1;
    }
    // Single wave in lockstep: just drain LDS before the cross-lane reload.
    asm volatile("s_wait_dscnt 0" ::: "memory");
    bf16x16 pa;
    {
      const int row = lane & 15;
      const int ko  = (lane & 16) >> 1;
      union { bf16x16 v; bf16x8 hh[2]; } u;
      u.hh[0] = *(const bf16x8*)(&Pt[row * 32 + ko]);
      u.hh[1] = *(const bf16x8*)(&Pt[row * 32 + ko + 16]);
      pa = u.v;
    }
#pragma unroll
    for (int t = 0; t < 4; ++t) {
#pragma unroll
      for (int i = 0; i < 8; ++i) o[t][i] *= alpha[i];
      const bf16x16 bv = load_tile_b(Vbh + (size_t)(t * 16) * SPAD + jp * 32, SPAD);
      o[t] = wmma_bf16(pa, bv, o[t]);
    }
  }

  float inv[8];
#pragma unroll
  for (int i = 0; i < 8; ++i) inv[i] = (lrow[i] > 0.f) ? 1.f / lrow[i] : 0.f;
#pragma unroll
  for (int t = 0; t < 4; ++t) {
#pragma unroll
    for (int i = 0; i < 8; ++i) {
      const int srow = qt * 16 + g8 + i;
      if (srow < SS)
        obuf[((size_t)b * SS + srow) * DD + h * DHH + t * 16 + n] = o[t][i] * inv[i];
    }
  }
}

// ---------------------------------------------------------------------------
// MLP GEMM1: h = gelu(yn @ w1 + b1).  M=18464, N=3072, K=768.
// One wave -> 64x64 output tile (16 accumulator tiles, 32 FLOP/byte).
// Tail M-block over-reads inside the workspace; stores are guarded.
// ---------------------------------------------------------------------------
__global__ __launch_bounds__(32) void mlp1_kernel(
    const __bf16* __restrict__ yn, const __bf16* __restrict__ w1t,
    const float* __restrict__ b1, __bf16* __restrict__ hbuf) {
  const int nt = blockIdx.x;                // N block of 64
  const int mt = blockIdx.y;                // M block of 64
  const int lane = threadIdx.x & 31;
  const int n  = lane & 15;
  const int g8 = (lane & 16) >> 1;
  const __bf16* arow = yn + (size_t)mt * 64 * DD;
  f32x8 acc[4][4] = {};
  for (int kk = 0; kk < DD; kk += 32) {
    bf16x16 a[4], bt[4];
#pragma unroll
    for (int mi = 0; mi < 4; ++mi)
      a[mi] = load_tile_a(arow + (size_t)mi * 16 * DD + kk, DD);
#pragma unroll
    for (int t = 0; t < 4; ++t)
      bt[t] = load_tile_b(w1t + (size_t)(nt * 64 + t * 16) * DD + kk, DD);
    if (kk + 32 < DD)   // hint the next B K-chunk (global_prefetch_b8)
      __builtin_prefetch(w1t + (size_t)nt * 64 * DD + kk + 32, 0, 1);
#pragma unroll
    for (int mi = 0; mi < 4; ++mi)
#pragma unroll
      for (int t = 0; t < 4; ++t)
        acc[mi][t] = wmma_bf16(a[mi], bt[t], acc[mi][t]);
  }
#pragma unroll
  for (int mi = 0; mi < 4; ++mi)
#pragma unroll
    for (int t = 0; t < 4; ++t) {
      const int col = nt * 64 + t * 16 + n;
      const float bias = b1[col];
#pragma unroll
      for (int i = 0; i < 8; ++i) {
        const int r = mt * 64 + mi * 16 + g8 + i;
        if (r < MROWS) {
          const float xv = acc[mi][t][i] + bias;
          const float gl = 0.5f * xv * (1.0f + erff(xv * 0.70710678118654752f));
          hbuf[(size_t)r * MLPD + col] = (__bf16)gl;
        }
      }
    }
}

// ---------------------------------------------------------------------------
// MLP GEMM2 + residual: y = out + h @ w2 + b2.  M=18464, N=768, K=3072.
// Same 64x64 register blocking.
// ---------------------------------------------------------------------------
__global__ __launch_bounds__(32) void mlp2_kernel(
    const __bf16* __restrict__ hbuf, const __bf16* __restrict__ w2t,
    const float* __restrict__ b2, const float* __restrict__ outb,
    float* __restrict__ y) {
  const int nt = blockIdx.x;                // N block of 64 (12 of them)
  const int mt = blockIdx.y;                // M block of 64
  const int lane = threadIdx.x & 31;
  const int n  = lane & 15;
  const int g8 = (lane & 16) >> 1;
  const __bf16* arow = hbuf + (size_t)mt * 64 * MLPD;
  f32x8 acc[4][4] = {};
  for (int kk = 0; kk < MLPD; kk += 32) {
    bf16x16 a[4], bt[4];
#pragma unroll
    for (int mi = 0; mi < 4; ++mi)
      a[mi] = load_tile_a(arow + (size_t)mi * 16 * MLPD + kk, MLPD);
#pragma unroll
    for (int t = 0; t < 4; ++t)
      bt[t] = load_tile_b(w2t + (size_t)(nt * 64 + t * 16) * MLPD + kk, MLPD);
    if (kk + 32 < MLPD)
      __builtin_prefetch(w2t + (size_t)nt * 64 * MLPD + kk + 32, 0, 1);
#pragma unroll
    for (int mi = 0; mi < 4; ++mi)
#pragma unroll
      for (int t = 0; t < 4; ++t)
        acc[mi][t] = wmma_bf16(a[mi], bt[t], acc[mi][t]);
  }
#pragma unroll
  for (int mi = 0; mi < 4; ++mi)
#pragma unroll
    for (int t = 0; t < 4; ++t) {
      const int col = nt * 64 + t * 16 + n;
      const float bias = b2[col];
#pragma unroll
      for (int i = 0; i < 8; ++i) {
        const size_t r = (size_t)mt * 64 + mi * 16 + g8 + i;
        if (r < (size_t)MROWS)
          y[r * DD + col] = outb[r * DD + col] + acc[mi][t][i] + bias;
      }
    }
}

// ---------------------------------------------------------------------------
// Host-side orchestration
// ---------------------------------------------------------------------------
extern "C" void kernel_launch(void* const* d_in, const int* in_sizes, int n_in,
                              void* d_out, int out_size, void* d_ws, size_t ws_size,
                              hipStream_t stream) {
  const float* x     = (const float*)d_in[0];
  const float* ln1_g = (const float*)d_in[1];
  const float* ln1_b = (const float*)d_in[2];
  const float* ln2_g = (const float*)d_in[3];
  const float* ln2_b = (const float*)d_in[4];
  const float* wq    = (const float*)d_in[5];
  const float* bq    = (const float*)d_in[6];
  const float* wk    = (const float*)d_in[7];
  const float* bk    = (const float*)d_in[8];
  const float* wv    = (const float*)d_in[9];
  const float* bv    = (const float*)d_in[10];
  const float* w1    = (const float*)d_in[11];
  const float* b1    = (const float*)d_in[12];
  const float* w2    = (const float*)d_in[13];
  const float* b2    = (const float*)d_in[14];
  float* y = (float*)d_out;

  char* wsp = (char*)d_ws;
  size_t off = 0;
  auto alloc = [&](size_t bytes) -> void* {
    void* p = wsp + off;
    off = (off + bytes + 255) & ~(size_t)255;
    return p;
  };

  __bf16* xn   = (__bf16*)alloc((size_t)BB * SPAD * DD * 2);
  __bf16* Qb   = (__bf16*)alloc((size_t)BB * HH * SPAD * DHH * 2);
  __bf16* Kb   = (__bf16*)alloc((size_t)BB * HH * SPAD * DHH * 2);
  __bf16* Vtb  = (__bf16*)alloc((size_t)BB * HH * DHH * SPAD * 2);
  float*  ob   = (float*) alloc((size_t)BB * SS * DD * 4);
  float*  outb = (float*) alloc((size_t)BB * SS * DD * 4);
  __bf16* yn   = (__bf16*)alloc((size_t)BB * SS * DD * 2);
  __bf16* hb   = (__bf16*)alloc((size_t)BB * SS * MLPD * 2);
  // Tail M-block of the MLP over-reads up to ~200KB past yn/hb; the weight
  // staging below provides in-workspace slack (stores are guarded).
  __bf16* wqT  = (__bf16*)alloc((size_t)HH * DHH * DHH * 2);
  __bf16* wkT  = (__bf16*)alloc((size_t)HH * DHH * DHH * 2);
  __bf16* wvT  = (__bf16*)alloc((size_t)HH * DHH * DHH * 2);
  __bf16* w1t  = (__bf16*)alloc((size_t)DD * MLPD * 2);
  __bf16* w2t  = (__bf16*)alloc((size_t)DD * MLPD * 2);
  (void)alloc(256 * 1024);   // slack so tail over-reads stay in-bounds

  // Weight pre-transpose/convert (B-operand wants N-major bf16).
  {
    const int tot = DHH * DHH;
    transpose_bf16_kernel<<<dim3((tot + 255) / 256, HH), 256, 0, stream>>>(wq, wqT, DHH, DHH);
    transpose_bf16_kernel<<<dim3((tot + 255) / 256, HH), 256, 0, stream>>>(wk, wkT, DHH, DHH);
    transpose_bf16_kernel<<<dim3((tot + 255) / 256, HH), 256, 0, stream>>>(wv, wvT, DHH, DHH);
    const int tot2 = DD * MLPD;
    transpose_bf16_kernel<<<dim3((tot2 + 255) / 256, 1), 256, 0, stream>>>(w1, w1t, DD, MLPD);
    transpose_bf16_kernel<<<dim3((tot2 + 255) / 256, 1), 256, 0, stream>>>(w2, w2t, MLPD, DD);
  }

  ln1_kernel<<<BB * SS, 256, 0, stream>>>(x, ln1_g, ln1_b, xn);

  qkv_kernel<<<dim3(STILES, HH, BB), 32, 0, stream>>>(
      xn, wqT, wkT, wvT, bq, bk, bv, Qb, Kb, Vtb);

  attn_kernel<<<dim3(STILES, HH, BB), 32, 0, stream>>>(Qb, Kb, Vtb, ob);

  resid_ln2_kernel<<<BB * SS, 256, 0, stream>>>(x, ob, ln2_g, ln2_b, outb, yn);

  mlp1_kernel<<<dim3(MLPD / 64, MT64), 32, 0, stream>>>(yn, w1t, b1, hb);

  mlp2_kernel<<<dim3(DD / 64, MT64), 32, 0, stream>>>(hb, w2t, b2, outb, y);
}
